// ESGP_73641509257884
// MI455X (gfx1250) — compile-verified, matrix operand order checked by reading, not run on hardware
//
#include <hip/hip_runtime.h>

#define BB 32
#define TT 1024
#define II 512
#define HH 1024

typedef __attribute__((ext_vector_type(16))) __bf16 v16bf;
typedef __attribute__((ext_vector_type(8)))  float  v8f;

union BV { v16bf v; uint4 q[2]; };

__device__ __forceinline__ __bf16 f2bf(float f) {
    union { float f; unsigned u; } uf; uf.f = f;
    unsigned r = uf.u + 0x7FFFu + ((uf.u >> 16) & 1u);   // round-to-nearest-even
    unsigned short s = (unsigned short)(r >> 16);
    union { unsigned short s; __bf16 b; } ub; ub.s = s;
    return ub.b;
}

// Load one wave's A/B fragment for v_wmma_f32_16x16x32_bf16 from a row-major
// bf16 matrix. Lane (p + 16q) supplies row (row), elements [c0..c0+7] and
// [c0+16..c0+23] where c0 = kk + 8q (ISA 7.12.2 16-bit A layout; B from a
// row-major W[h_out,k] weight matrix loads identically).
__device__ __forceinline__ v16bf load_bf_frag(const __bf16* base, int row,
                                              long long ld, int c0) {
    BV bv;
    const __bf16* p = base + (size_t)row * (size_t)ld + c0;
    bv.q[0] = *reinterpret_cast<const uint4*>(p);
    bv.q[1] = *reinterpret_cast<const uint4*>(p + 16);
    return bv.v;
}

// One recurrent step for one layer (all operands bf16, f32 accumulate).
//   z      = sigmoid(A1 @ Wz_a1^T + A2 @ Wz_a2^T + bias)
//   h~     = tanh   (A1 @ Wh_a1^T + A2 @ Wh_a2^T)
//   h_new  = (1-z)*h_prev + z*h~
// A1: layer input (bf16 x_t for layer 0, bf16 h_l0 for layer 1), A2: bf16 h_prev.
// 128 waves: wave -> (m_tile in {0,1}, n_tile in 0..63); each wave owns one
// 16x16 output tile and both accumulators (2 WMMAs per 32-wide K step,
// A fragment reused).
__global__ __launch_bounds__(256) void esn_step_kernel(
    const __bf16* __restrict__ a1_bf, long long a1_ld, int K1,
    const __bf16* __restrict__ a2_bf, int K2,
    const __bf16* __restrict__ Wz_a1, const __bf16* __restrict__ Wz_a2,
    const __bf16* __restrict__ Wh_a1, const __bf16* __restrict__ Wh_a2,
    const float* __restrict__ bias,
    const float* __restrict__ h_prev,
    float* __restrict__ h_new, __bf16* __restrict__ h_new_bf,
    float* __restrict__ out_ptr, long long out_ld)
{
    const int lane = threadIdx.x & 31;
    const int wave = blockIdx.x * (blockDim.x >> 5) + (threadIdx.x >> 5);
    const int m0 = (wave & 1) << 4;
    const int n0 = (wave >> 1) << 4;
    const int p = lane & 15;
    const int q = lane >> 4;
    const int arow = m0 + p;   // A-operand row this lane feeds
    const int brow = n0 + p;   // weight row (= output column) this lane feeds

    v8f accz = {};
    v8f acch = {};

    for (int kk = 0; kk < K1; kk += 32) {
        const int c0 = kk + (q << 3);
        v16bf a  = load_bf_frag(a1_bf, arow, a1_ld, c0);
        v16bf bz = load_bf_frag(Wz_a1, brow, K1, c0);
        accz = __builtin_amdgcn_wmma_f32_16x16x32_bf16(false, a, false, bz,
                                                       (short)0, accz, false, false);
        v16bf bh = load_bf_frag(Wh_a1, brow, K1, c0);
        acch = __builtin_amdgcn_wmma_f32_16x16x32_bf16(false, a, false, bh,
                                                       (short)0, acch, false, false);
    }
    for (int kk = 0; kk < K2; kk += 32) {
        const int c0 = kk + (q << 3);
        v16bf a  = load_bf_frag(a2_bf, arow, K2, c0);
        v16bf bz = load_bf_frag(Wz_a2, brow, K2, c0);
        accz = __builtin_amdgcn_wmma_f32_16x16x32_bf16(false, a, false, bz,
                                                       (short)0, accz, false, false);
        v16bf bh = load_bf_frag(Wh_a2, brow, K2, c0);
        acch = __builtin_amdgcn_wmma_f32_16x16x32_bf16(false, a, false, bh,
                                                       (short)0, acch, false, false);
    }

    // C/D layout: VGPR r, lanes 0-15 -> M = m0+r, lanes 16-31 -> M = m0+8+r,
    // N = n0 + (lane&15) for both halves.
    const int ncol = n0 + p;
    const float bv = bias ? bias[ncol] : 0.0f;
#pragma unroll
    for (int r = 0; r < 8; ++r) {
        const int row = m0 + (q << 3) + r;
        const float zpre = accz[r] + bv;
        const float z = 1.0f / (1.0f + __expf(-zpre));
        const float e2 = __expf(-2.0f * acch[r]);
        const float ht = (1.0f - e2) / (1.0f + e2);
        const float hp = h_prev[(size_t)row * HH + ncol];
        const float hn = (1.0f - z) * hp + z * ht;
        h_new[(size_t)row * HH + ncol] = hn;
        h_new_bf[(size_t)row * HH + ncol] = f2bf(hn);
        if (out_ptr)
            out_ptr[(size_t)row * (size_t)out_ld + ncol] = hn;
    }
}

// Vectorized fp32 -> bf16 conversion: 4 elements per thread (n % 4 == 0).
__global__ void cvt4_f32_bf16_kernel(const float* __restrict__ src,
                                     __bf16* __restrict__ dst, int n4) {
    int i = blockIdx.x * blockDim.x + threadIdx.x;
    if (i < n4) {
        float4 f = reinterpret_cast<const float4*>(src)[i];
        union { __bf16 b[4]; uint2 u; } o;
        o.b[0] = f2bf(f.x); o.b[1] = f2bf(f.y);
        o.b[2] = f2bf(f.z); o.b[3] = f2bf(f.w);
        reinterpret_cast<uint2*>(dst)[i] = o.u;
    }
}

__global__ void init_h_kernel(const float* __restrict__ h0,
                              float* __restrict__ h0f, float* __restrict__ h1f,
                              __bf16* __restrict__ h0b, __bf16* __restrict__ h1b) {
    int i = blockIdx.x * blockDim.x + threadIdx.x;
    if (i < BB * HH) {
        float a = h0[i];
        float b = h0[BB * HH + i];
        h0f[i] = a; h1f[i] = b;
        h0b[i] = f2bf(a); h1b[i] = f2bf(b);
    }
}

__global__ void finalize_kernel(const float* __restrict__ h0f,
                                const float* __restrict__ h1f,
                                float* __restrict__ hn) {
    int i = blockIdx.x * blockDim.x + threadIdx.x;
    if (i < BB * HH) {
        hn[i] = h0f[i];
        hn[BB * HH + i] = h1f[i];
    }
}

extern "C" void kernel_launch(void* const* d_in, const int* in_sizes, int n_in,
                              void* d_out, int out_size, void* d_ws, size_t ws_size,
                              hipStream_t stream) {
    (void)in_sizes; (void)n_in; (void)out_size; (void)ws_size;
    const float* x      = (const float*)d_in[0];   // (B,T,I)
    const float* h0     = (const float*)d_in[1];   // (L,B,H)
    const float* W_in0  = (const float*)d_in[2];   // (H,I)
    const float* W_res0 = (const float*)d_in[3];   // (H,H)
    const float* W_z0   = (const float*)d_in[4];   // (H,I)
    const float* U_z0   = (const float*)d_in[5];   // (H,H)
    const float* b_z0   = (const float*)d_in[6];   // (H)
    const float* W_in1  = (const float*)d_in[7];   // (H,H)
    const float* W_res1 = (const float*)d_in[8];   // (H,H)
    const float* W_z1   = (const float*)d_in[9];   // (H,H)
    const float* U_z1   = (const float*)d_in[10];  // (H,H)
    const float* b_z1   = (const float*)d_in[11];  // (H)
    float* out = (float*)d_out;

    // ---- carve workspace (~50 MB) ----
    char* w = (char*)d_ws;
    auto alloc_bf = [&](size_t n) { __bf16* p = (__bf16*)w; w += n * sizeof(__bf16); return p; };
    auto alloc_f  = [&](size_t n) { float*  p = (float*)w;  w += n * sizeof(float);  return p; };
    __bf16* x_bf   = alloc_bf((size_t)BB * TT * II);   // bf16 copy of x
    __bf16* Wz0b   = alloc_bf((size_t)HH * II);
    __bf16* Win0b  = alloc_bf((size_t)HH * II);
    __bf16* Uz0b   = alloc_bf((size_t)HH * HH);
    __bf16* Wres0b = alloc_bf((size_t)HH * HH);
    __bf16* Wz1b   = alloc_bf((size_t)HH * HH);
    __bf16* Uz1b   = alloc_bf((size_t)HH * HH);
    __bf16* Win1b  = alloc_bf((size_t)HH * HH);
    __bf16* Wres1b = alloc_bf((size_t)HH * HH);
    float*  hl0f[2] = { alloc_f((size_t)BB * HH), alloc_f((size_t)BB * HH) };
    float*  hl1f[2] = { alloc_f((size_t)BB * HH), alloc_f((size_t)BB * HH) };
    __bf16* hl0b[2] = { alloc_bf((size_t)BB * HH), alloc_bf((size_t)BB * HH) };
    __bf16* hl1b[2] = { alloc_bf((size_t)BB * HH), alloc_bf((size_t)BB * HH) };

    // ---- one-time (per call) conversions: x and weights to bf16 ----
    auto cvt = [&](const float* s, __bf16* d, size_t n) {
        int n4 = (int)(n / 4);
        cvt4_f32_bf16_kernel<<<(n4 + 255) / 256, 256, 0, stream>>>(s, d, n4);
    };
    cvt(x,      x_bf,   (size_t)BB * TT * II);
    cvt(W_z0,   Wz0b,   (size_t)HH * II);
    cvt(W_in0,  Win0b,  (size_t)HH * II);
    cvt(U_z0,   Uz0b,   (size_t)HH * HH);
    cvt(W_res0, Wres0b, (size_t)HH * HH);
    cvt(W_z1,   Wz1b,   (size_t)HH * HH);
    cvt(U_z1,   Uz1b,   (size_t)HH * HH);
    cvt(W_in1,  Win1b,  (size_t)HH * HH);
    cvt(W_res1, Wres1b, (size_t)HH * HH);
    init_h_kernel<<<(BB * HH + 255) / 256, 256, 0, stream>>>(
        h0, hl0f[0], hl1f[0], hl0b[0], hl1b[0]);

    // ---- sequential recurrence: 2 kernels per timestep, stream-ordered ----
    dim3 grid(16), block(256);   // 128 waves = 2 M-tiles x 64 N-tiles
    for (int t = 0; t < TT; ++t) {
        const int rd = t & 1, wr = (t + 1) & 1;
        // layer 0: A1 = x_t (bf16, row stride T*I), A2 = h_l0 prev (bf16)
        esn_step_kernel<<<grid, block, 0, stream>>>(
            x_bf + (size_t)t * II, (long long)TT * II, II,
            hl0b[rd], HH,
            Wz0b, Uz0b, Win0b, Wres0b,
            b_z0, hl0f[rd], hl0f[wr], hl0b[wr],
            (float*)nullptr, (long long)0);
        // layer 1: A1 = h_l0 new (bf16), A2 = h_l1 prev (bf16); writes output[:,t,:]
        esn_step_kernel<<<grid, block, 0, stream>>>(
            hl0b[wr], (long long)HH, HH,
            hl1b[rd], HH,
            Wz1b, Uz1b, Win1b, Wres1b,
            b_z1, hl1f[rd], hl1f[wr], hl1b[wr],
            out + (size_t)t * HH, (long long)TT * HH);
    }
    // final hidden states live in ping-pong slot (T % 2) == 0
    finalize_kernel<<<(BB * HH + 255) / 256, 256, 0, stream>>>(
        hl0f[0], hl1f[0], out + (size_t)BB * TT * HH);
}
